// ICLBlock_67516885893231
// MI455X (gfx1250) — matrix operand order repaired and checked by reading, not compiled, not run on hardware
//
#include <hip/hip_runtime.h>
#include <math.h>

// ---------------------------------------------------------------------------
// MI455X / gfx1250: all dense math via v_wmma_f32_16x16x32_f16 (wave32),
// f32 accumulation, flash attention with online softmax, double-buffered
// async (global -> LDS) weight staging in the GEMMs.
// ---------------------------------------------------------------------------

typedef __attribute__((ext_vector_type(16))) _Float16 v16h;
typedef __attribute__((ext_vector_type(8)))  _Float16 v8h;
typedef __attribute__((ext_vector_type(8)))  float    v8f;
typedef __attribute__((ext_vector_type(4)))  int      v4i;

#define WMMA16(a, b, c) \
  __builtin_amdgcn_wmma_f32_16x16x32_f16(false, (a), false, (b), (short)0, (c), false, false)

// ---- CDNA5 async global->LDS path (guarded; plain-copy fallback) ----------
#if defined(__HIP_DEVICE_COMPILE__) && __has_builtin(__builtin_amdgcn_global_load_async_to_lds_b128)
#define HAVE_ASYNC_LDS 1
#else
#define HAVE_ASYNC_LDS 0
#endif

#if HAVE_ASYNC_LDS
typedef __attribute__((address_space(1))) v4i* g_v4i_p;  // global src
typedef __attribute__((address_space(3))) v4i* l_v4i_p;  // LDS dst
#define ASYNC_CP16(srcp, dstp)                              \
  __builtin_amdgcn_global_load_async_to_lds_b128(           \
      (g_v4i_p)(uintptr_t)(srcp),                           \
      (l_v4i_p)(unsigned)(uintptr_t)(dstp), 0, 0)
#if __has_builtin(__builtin_amdgcn_s_wait_asynccnt)
#define WAIT_ASYNC(n) __builtin_amdgcn_s_wait_asynccnt(n)
#else
#define WAIT_ASYNC(n) asm volatile("s_wait_asynccnt %0" ::"i"(n) : "memory")
#endif
#else
#define ASYNC_CP16(srcp, dstp) (*(v8h*)(dstp) = *(const v8h*)(srcp))
#define WAIT_ASYNC(n)
#endif

static constexpr int B_    = 2;
static constexpr int S_    = 2048;
static constexpr int E_    = 1024;
static constexpr int H_    = 8;
static constexpr int DH_   = 64;
static constexpr int DA_   = 512;   // H*DH
static constexpr int DMLP_ = 2048;  // 4*(E/2)
static constexpr int M_    = B_ * S_;          // 4096 rows
static constexpr float SCALE_  = 0.044194173824159216f; // 1/sqrt(DA)
static constexpr float LN_EPS_ = 1e-5f;

// ---------------------------------------------------------------------------
// Convert f32 weight W[K,N] -> f16 transposed Wt[N,K] (B-fragment friendly:
// per-lane 32B contiguous-K loads).
// ---------------------------------------------------------------------------
__global__ void wt_convert(const float* __restrict__ W, _Float16* __restrict__ Wt,
                           int K, int N) {
  int idx = blockIdx.x * 256 + threadIdx.x;
  if (idx >= K * N) return;
  int k = idx / N;
  int n = idx - k * N;
  Wt[(size_t)n * K + k] = (_Float16)W[idx];
}

// ---------------------------------------------------------------------------
// Row LayerNorm over E=1024, f32 in -> f16 out. One 256-thread block per row.
// ---------------------------------------------------------------------------
__global__ void ln_f16(const float* __restrict__ x, const float* __restrict__ g,
                       const float* __restrict__ bb, _Float16* __restrict__ out) {
  __shared__ float s1[256];
  __shared__ float s2[256];
  const int row = blockIdx.x;
  const int tid = threadIdx.x;
  const float* xr = x + (size_t)row * E_;
  float4 v = ((const float4*)xr)[tid];
  float sum = v.x + v.y + v.z + v.w;
  float sq  = v.x * v.x + v.y * v.y + v.z * v.z + v.w * v.w;
  s1[tid] = sum; s2[tid] = sq;
  __syncthreads();
  for (int off = 128; off > 0; off >>= 1) {
    if (tid < off) { s1[tid] += s1[tid + off]; s2[tid] += s2[tid + off]; }
    __syncthreads();
  }
  const float mu  = s1[0] * (1.0f / E_);
  const float var = s2[0] * (1.0f / E_) - mu * mu;
  const float inv = rsqrtf(var + LN_EPS_);
  const int i = tid * 4;
  _Float16* orow = out + (size_t)row * E_;
  orow[i + 0] = (_Float16)((v.x - mu) * inv * g[i + 0] + bb[i + 0]);
  orow[i + 1] = (_Float16)((v.y - mu) * inv * g[i + 1] + bb[i + 1]);
  orow[i + 2] = (_Float16)((v.z - mu) * inv * g[i + 2] + bb[i + 2]);
  orow[i + 3] = (_Float16)((v.w - mu) * inv * g[i + 3] + bb[i + 3]);
}

// ---------------------------------------------------------------------------
// WMMA GEMM: C[M,N] = epilogue(A[M,K] @ Wt[N,K]^T).
//   MODE 0: outh = f16( gelu_exact(acc + bias) )                (MLP fc1)
//   MODE 1: outf = res + acc (+bias);  optional f16 copy outh   (fc2 / wo)
//   MODE 2: outh scattered to [B,H,S,DH] head layout            (q/k/v proj)
// Block: 128 threads (4 waves). WG tile 64x64; each wave 16(M) x 64(N).
// Wt k-tiles (64x32 f16 = 4KB) are double-buffered in LDS via async
// global->LDS copies shared by all 4 waves; A rows load direct from global.
// ---------------------------------------------------------------------------
template <int MODE>
__global__ void gemm_f16(const _Float16* __restrict__ A,
                         const _Float16* __restrict__ Wt,
                         const float* __restrict__ bias,
                         const float* res, float* outf,
                         _Float16* __restrict__ outh,
                         int M, int N, int K) {
  __shared__ __align__(64) _Float16 wbuf[2][64 * 32];
  const int tid  = threadIdx.x;
  const int wave = tid >> 5;
  const int lane = tid & 31;
  const int li = lane & 15;
  const int hi = lane >> 4;
  const int m0 = blockIdx.y * 64 + wave * 16;
  const int n0 = blockIdx.x * 64;
  // staging map: 128 threads x 2 chunks of 16B cover 64 rows x 64B
  const int srow  = tid >> 1;        // 0..63
  const int spart = (tid & 1) << 1;  // {0,2} 16B-part index

  const _Float16* arow  = A + (size_t)(m0 + li) * K;
  const _Float16* wsrc0 = Wt + (size_t)(n0 + srow) * K + spart * 8;
  _Float16* wdst0 = &wbuf[0][srow * 32 + spart * 8];
  _Float16* wdst1 = &wbuf[1][srow * 32 + spart * 8];

  v8f acc[4];
  const v8f zero = {0.f, 0.f, 0.f, 0.f, 0.f, 0.f, 0.f, 0.f};
#pragma unroll
  for (int nt = 0; nt < 4; ++nt) acc[nt] = zero;

  const int nk = K >> 5;
  // prologue: stage k-tile 0 into buffer 0
  ASYNC_CP16(wsrc0, wdst0);
  ASYNC_CP16(wsrc0 + 8, wdst0 + 8);

  for (int kt = 0; kt < nk; ++kt) {
    const int k0 = kt << 5;
    const int cur = kt & 1;
    const bool more = (kt + 1) < nk;
    if (more) {  // stage next tile into the other buffer (overlaps compute)
      const _Float16* s = wsrc0 + k0 + 32;
      _Float16* d = (cur == 0) ? wdst1 : wdst0;
      ASYNC_CP16(s, d);
      ASYNC_CP16(s + 8, d + 8);
    }
#if HAVE_ASYNC_LDS
    if (more) { WAIT_ASYNC(2); } else { WAIT_ASYNC(0); }  // current tile retired
#endif
    __syncthreads();  // cross-wave visibility of wbuf[cur]
    if (more) __builtin_prefetch(arow + k0 + 32, 0, 0);   // global_prefetch_b8

    v8h alo = *(const v8h*)(arow + k0 + 16 * hi);
    v8h ahi = *(const v8h*)(arow + k0 + 16 * hi + 8);
    v16h a = __builtin_shufflevector(alo, ahi, 0, 1, 2, 3, 4, 5, 6, 7,
                                     8, 9, 10, 11, 12, 13, 14, 15);
#pragma unroll
    for (int nt = 0; nt < 4; ++nt) {
      v16h b = *(const v16h*)&wbuf[cur][(nt * 16 + li) * 32 + 16 * hi];
      acc[nt] = WMMA16(a, b, acc[nt]);
    }
    __syncthreads();  // all waves done reading before buffer is re-staged
  }

#pragma unroll
  for (int nt = 0; nt < 4; ++nt) {
    const int n = n0 + nt * 16 + li;
#pragma unroll
    for (int r = 0; r < 8; ++r) {
      const int m = m0 + 8 * hi + r;
      float z = acc[nt][r];
      if (MODE == 0) {
        z += bias[n];
        float gl = 0.5f * z * (1.0f + erff(z * 0.70710678118654752f));
        outh[(size_t)m * N + n] = (_Float16)gl;
      } else if (MODE == 1) {
        if (bias) z += bias[n];
        float f = res[(size_t)m * N + n] + z;
        outf[(size_t)m * N + n] = f;
        if (outh) outh[(size_t)m * N + n] = (_Float16)f;
      } else {  // MODE 2: scatter (m=b*S+s, n=h*DH+dh) -> [b,h,s,dh]
        const int h  = n >> 6;
        const int dh = n & 63;
        const int b  = m >> 11;       // S_ = 2048
        const int s  = m & 2047;
        outh[(((size_t)b * H_ + h) * S_ + s) * DH_ + dh] = (_Float16)z;
      }
    }
  }
}

// ---------------------------------------------------------------------------
// Interleaved RoPE in-place on f16 [B*H, S, DH]. One thread per (bh,s,pair).
// ---------------------------------------------------------------------------
__global__ void rope_f16(_Float16* __restrict__ x) {
  const int idx = blockIdx.x * 256 + threadIdx.x;  // B*H*S*32 total
  const int i  = idx & 31;
  const int s  = (idx >> 5) & (S_ - 1);
  const int bh = idx >> 16;
  _Float16* p = x + ((size_t)bh * S_ + s) * DH_ + 2 * i;
  const float inv_freq = powf(10000.0f, -(float)(2 * i) / (float)DH_);
  const float ang = (float)s * inv_freq;
  const float cs = cosf(ang), sn = sinf(ang);
  const float x1 = (float)p[0], x2 = (float)p[1];
  p[0] = (_Float16)(x1 * cs - x2 * sn);
  p[1] = (_Float16)(x1 * sn + x2 * cs);
}

// ---------------------------------------------------------------------------
// Flash attention, one wave (32 threads) per (bh, 16-query-row block).
// QK^T and P*V via WMMA; online softmax with shfl reductions; P transposed
// C-layout -> A-layout through LDS; V tile staged transposed in LDS.
// ICL mask: drop t >= s except (0,0).  COV mask: drop t > s (causal).
// ---------------------------------------------------------------------------
template <bool ICL>
__global__ void attn_f16(const _Float16* __restrict__ qp,
                         const _Float16* __restrict__ kp,
                         const _Float16* __restrict__ vp,
                         _Float16* __restrict__ out) {
  __shared__ __align__(64) _Float16 vt[64 * 32];     // V^T tile: [dh][t]
  __shared__ __align__(64) _Float16 ptile[16 * 32];  // P tile:   [s][t]

  const int bh = blockIdx.y;
  const int s0 = blockIdx.x * 16;
  const int lane = threadIdx.x;
  const int li = lane & 15;
  const int hi = lane >> 4;
  const size_t base = (size_t)bh * S_ * DH_;
  const float NEG_INF = -__builtin_inff();

  // Q A-fragments for this 16-row block (DH=64 -> two K=32 fragments)
  v16h aq[2];
  {
    const _Float16* qrow = qp + base + (size_t)(s0 + li) * DH_;
#pragma unroll
    for (int d = 0; d < 2; ++d) {
      v8h lo  = *(const v8h*)(qrow + d * 32 + 16 * hi);
      v8h hi8 = *(const v8h*)(qrow + d * 32 + 16 * hi + 8);
      aq[d] = __builtin_shufflevector(lo, hi8, 0, 1, 2, 3, 4, 5, 6, 7,
                                      8, 9, 10, 11, 12, 13, 14, 15);
    }
  }

  float Mx[8], Ls[8];
  v8f oacc[4];
  const v8f zero = {0.f, 0.f, 0.f, 0.f, 0.f, 0.f, 0.f, 0.f};
#pragma unroll
  for (int r = 0; r < 8; ++r) { Mx[r] = -1e30f; Ls[r] = 0.0f; }
#pragma unroll
  for (int d = 0; d < 4; ++d) oacc[d] = zero;

  for (int t0 = 0; t0 < s0 + 16; t0 += 32) {
    __syncthreads();  // previous iteration finished reading vt/ptile
    // Stage V tile transposed: lane loads V row (t0+lane), writes column.
    {
      const v16h* vr = (const v16h*)(vp + base + (size_t)(t0 + lane) * DH_);
#pragma unroll
      for (int c = 0; c < 4; ++c) {
        v16h chunk = vr[c];
#pragma unroll
        for (int j = 0; j < 16; ++j) vt[(c * 16 + j) * 32 + lane] = chunk[j];
      }
    }

    // Scores: two 16x16 tiles covering t in [t0, t0+32)
    v8f sc[2];
#pragma unroll
    for (int tile = 0; tile < 2; ++tile) {
      v8f c = zero;
      const int t = t0 + tile * 16 + li;
      const _Float16* krow = kp + base + (size_t)t * DH_;
#pragma unroll
      for (int d = 0; d < 2; ++d) {
        v16h b = *(const v16h*)(krow + d * 32 + 16 * hi);
        c = WMMA16(aq[d], b, c);
      }
      sc[tile] = c;
    }

    // Online softmax update + build P tile in LDS (f16).
#pragma unroll
    for (int r = 0; r < 8; ++r) {
      const int srow = s0 + 8 * hi + r;
      float x0 = sc[0][r] * SCALE_;
      float x1 = sc[1][r] * SCALE_;
      {
        const int t = t0 + li;
        const bool mk = ICL ? (t >= srow && !(srow == 0 && t == 0)) : (t > srow);
        if (mk) x0 = NEG_INF;
      }
      {
        const int t = t0 + 16 + li;
        const bool mk = ICL ? (t >= srow && !(srow == 0 && t == 0)) : (t > srow);
        if (mk) x1 = NEG_INF;
      }
      float mnew = fmaxf(x0, x1);
#pragma unroll
      for (int off = 1; off < 16; off <<= 1)
        mnew = fmaxf(mnew, __shfl_xor(mnew, off, 32));
      const float Mn = fmaxf(Mx[r], mnew);       // stays finite (init -1e30)
      const float corr = __expf(Mx[r] - Mn);
      Mx[r] = Mn;
      const float p0 = __expf(x0 - Mn);
      const float p1 = __expf(x1 - Mn);
      float rs = p0 + p1;
#pragma unroll
      for (int off = 1; off < 16; off <<= 1)
        rs += __shfl_xor(rs, off, 32);
      Ls[r] = Ls[r] * corr + rs;
#pragma unroll
      for (int d = 0; d < 4; ++d) oacc[d][r] *= corr;
      ptile[(8 * hi + r) * 32 + li]      = (_Float16)p0;
      ptile[(8 * hi + r) * 32 + 16 + li] = (_Float16)p1;
    }
    __syncthreads();  // vt + ptile visible across lanes

    // O += P @ V  (A from ptile, B from transposed V tile, K=32)
    v16h pa;
    {
      v8h lo  = *(const v8h*)(ptile + li * 32 + 16 * hi);
      v8h hi8 = *(const v8h*)(ptile + li * 32 + 16 * hi + 8);
      pa = __builtin_shufflevector(lo, hi8, 0, 1, 2, 3, 4, 5, 6, 7,
                                   8, 9, 10, 11, 12, 13, 14, 15);
    }
#pragma unroll
    for (int d = 0; d < 4; ++d) {
      v16h b = *(const v16h*)(vt + (d * 16 + li) * 32 + 16 * hi);
      oacc[d] = WMMA16(pa, b, oacc[d]);
    }
  }

  // Epilogue: normalize and gather back to [b, s, h*DH + dh] (f16 for @Wo).
  const int b = bh >> 3;   // H_ = 8
  const int h = bh & 7;
#pragma unroll
  for (int d = 0; d < 4; ++d) {
#pragma unroll
    for (int r = 0; r < 8; ++r) {
      const int s = s0 + 8 * hi + r;
      const int col = h * DH_ + d * 16 + li;
      const float vv = oacc[d][r] / Ls[r];
      out[((size_t)b * S_ + s) * DA_ + col] = (_Float16)vv;
    }
  }
}

// ---------------------------------------------------------------------------
// Orchestration
// ---------------------------------------------------------------------------
extern "C" void kernel_launch(void* const* d_in, const int* in_sizes, int n_in,
                              void* d_out, int out_size, void* d_ws, size_t ws_size,
                              hipStream_t stream) {
  (void)in_sizes; (void)n_in; (void)out_size; (void)ws_size;

  const float* covariates = (const float*)d_in[0];
  const float* targets    = (const float*)d_in[1];
  const float* func_up    = (const float*)d_in[2];
  const float* fc1_w = (const float*)d_in[3];
  const float* fc1_b = (const float*)d_in[4];
  const float* fc2_w = (const float*)d_in[5];
  const float* fc2_b = (const float*)d_in[6];
  const float* icl_wq = (const float*)d_in[7];
  const float* icl_wk = (const float*)d_in[8];
  const float* icl_wv = (const float*)d_in[9];
  const float* icl_wo = (const float*)d_in[10];
  const float* cov_wq = (const float*)d_in[11];
  const float* cov_wk = (const float*)d_in[12];
  const float* cov_wv = (const float*)d_in[13];
  const float* cov_wo = (const float*)d_in[14];
  const float* ln_cm_g = (const float*)d_in[15];
  const float* ln_cm_b = (const float*)d_in[16];
  const float* ln_ia_g = (const float*)d_in[17];
  const float* ln_ia_b = (const float*)d_in[18];
  const float* ln_im_g = (const float*)d_in[19];
  const float* ln_im_b = (const float*)d_in[20];

  float* out_cov = (float*)d_out;
  float* out_tgt = out_cov + (size_t)M_ * E_;
  float* out_fu  = out_tgt + (size_t)M_ * E_;

  char* ws = (char*)d_ws;
  size_t off = 0;
  auto alloc = [&](size_t elems) -> _Float16* {
    _Float16* p = (_Float16*)(ws + off);
    off += ((elems * sizeof(_Float16)) + 255) & ~(size_t)255;
    return p;
  };
  _Float16* wt_fc1 = alloc((size_t)E_ * DMLP_);   // [DMLP][E]
  _Float16* wt_fc2 = alloc((size_t)DMLP_ * E_);   // [E][DMLP]
  _Float16* wt_iq  = alloc((size_t)E_ * DA_);     // [DA][E]
  _Float16* wt_ik  = alloc((size_t)E_ * DA_);
  _Float16* wt_iv  = alloc((size_t)E_ * DA_);
  _Float16* wt_io  = alloc((size_t)DA_ * E_);     // [E][DA]
  _Float16* wt_cq  = alloc((size_t)E_ * DA_);
  _Float16* wt_ck  = alloc((size_t)E_ * DA_);
  _Float16* wt_cv  = alloc((size_t)E_ * DA_);
  _Float16* wt_co  = alloc((size_t)DA_ * E_);
  _Float16* lnbuf  = alloc((size_t)M_ * E_);
  _Float16* hbuf   = alloc((size_t)M_ * DMLP_);
  _Float16* qbuf   = alloc((size_t)M_ * E_);
  _Float16* tgtbuf = alloc((size_t)M_ * E_);
  _Float16* qp     = alloc((size_t)M_ * DA_);     // [B,H,S,DH]
  _Float16* kp     = alloc((size_t)M_ * DA_);
  _Float16* vp     = alloc((size_t)M_ * DA_);
  _Float16* aout   = alloc((size_t)M_ * DA_);     // [B,S,DA]

  auto cdiv = [](int a, int b) { return (a + b - 1) / b; };

  // --- weight conversion (f32 -> f16, transposed) ---
  wt_convert<<<cdiv(E_ * DMLP_, 256), 256, 0, stream>>>(fc1_w, wt_fc1, E_, DMLP_);
  wt_convert<<<cdiv(DMLP_ * E_, 256), 256, 0, stream>>>(fc2_w, wt_fc2, DMLP_, E_);
  wt_convert<<<cdiv(E_ * DA_, 256), 256, 0, stream>>>(icl_wq, wt_iq, E_, DA_);
  wt_convert<<<cdiv(E_ * DA_, 256), 256, 0, stream>>>(icl_wk, wt_ik, E_, DA_);
  wt_convert<<<cdiv(E_ * DA_, 256), 256, 0, stream>>>(icl_wv, wt_iv, E_, DA_);
  wt_convert<<<cdiv(DA_ * E_, 256), 256, 0, stream>>>(icl_wo, wt_io, DA_, E_);
  wt_convert<<<cdiv(E_ * DA_, 256), 256, 0, stream>>>(cov_wq, wt_cq, E_, DA_);
  wt_convert<<<cdiv(E_ * DA_, 256), 256, 0, stream>>>(cov_wk, wt_ck, E_, DA_);
  wt_convert<<<cdiv(E_ * DA_, 256), 256, 0, stream>>>(cov_wv, wt_cv, E_, DA_);
  wt_convert<<<cdiv(DA_ * E_, 256), 256, 0, stream>>>(cov_wo, wt_co, DA_, E_);

  const dim3 blk128(128);
  const dim3 g_fc1(DMLP_ / 64, M_ / 64);
  const dim3 g_fc2(E_ / 64, M_ / 64);
  const dim3 g_prj(DA_ / 64, M_ / 64);
  const dim3 g_att(S_ / 16, B_ * H_);
  const int rope_blocks = (B_ * H_ * S_ * 32) / 256;

  // --- cov = covariates + MLP(LN_cm(covariates)) ---
  ln_f16<<<M_, 256, 0, stream>>>(covariates, ln_cm_g, ln_cm_b, lnbuf);
  gemm_f16<0><<<g_fc1, blk128, 0, stream>>>(lnbuf, wt_fc1, fc1_b, nullptr, nullptr,
                                            hbuf, M_, DMLP_, E_);
  gemm_f16<1><<<g_fc2, blk128, 0, stream>>>(hbuf, wt_fc2, fc2_b, covariates, out_cov,
                                            nullptr, M_, E_, DMLP_);

  // --- tgt = targets + MLP(LN_im(functional_update)) ---
  ln_f16<<<M_, 256, 0, stream>>>(func_up, ln_im_g, ln_im_b, lnbuf);
  gemm_f16<0><<<g_fc1, blk128, 0, stream>>>(lnbuf, wt_fc1, fc1_b, nullptr, nullptr,
                                            hbuf, M_, DMLP_, E_);
  gemm_f16<1><<<g_fc2, blk128, 0, stream>>>(hbuf, wt_fc2, fc2_b, targets, out_tgt,
                                            tgtbuf, M_, E_, DMLP_);

  // --- q = LN_ia(cov) ---
  ln_f16<<<M_, 256, 0, stream>>>(out_cov, ln_ia_g, ln_ia_b, qbuf);

  // --- ICL attention: fu = func_up + attn(q, q, tgt) @ icl_wo ---
  gemm_f16<2><<<g_prj, blk128, 0, stream>>>(qbuf, wt_iq, nullptr, nullptr, nullptr,
                                            qp, M_, DA_, E_);
  gemm_f16<2><<<g_prj, blk128, 0, stream>>>(qbuf, wt_ik, nullptr, nullptr, nullptr,
                                            kp, M_, DA_, E_);
  gemm_f16<2><<<g_prj, blk128, 0, stream>>>(tgtbuf, wt_iv, nullptr, nullptr, nullptr,
                                            vp, M_, DA_, E_);
  rope_f16<<<rope_blocks, 256, 0, stream>>>(qp);
  rope_f16<<<rope_blocks, 256, 0, stream>>>(kp);
  attn_f16<true><<<g_att, 32, 0, stream>>>(qp, kp, vp, aout);
  gemm_f16<1><<<g_fc2, blk128, 0, stream>>>(aout, wt_io, nullptr, func_up, out_fu,
                                            nullptr, M_, E_, DA_);

  // --- COV attention: cov += attn(q, q, q) @ cov_wo ---
  gemm_f16<2><<<g_prj, blk128, 0, stream>>>(qbuf, wt_cq, nullptr, nullptr, nullptr,
                                            qp, M_, DA_, E_);
  gemm_f16<2><<<g_prj, blk128, 0, stream>>>(qbuf, wt_ck, nullptr, nullptr, nullptr,
                                            kp, M_, DA_, E_);
  gemm_f16<2><<<g_prj, blk128, 0, stream>>>(qbuf, wt_cv, nullptr, nullptr, nullptr,
                                            vp, M_, DA_, E_);
  rope_f16<<<rope_blocks, 256, 0, stream>>>(qp);
  rope_f16<<<rope_blocks, 256, 0, stream>>>(kp);
  attn_f16<false><<<g_att, 32, 0, stream>>>(qp, kp, vp, aout);
  gemm_f16<1><<<g_fc2, blk128, 0, stream>>>(aout, wt_co, nullptr, out_cov, out_cov,
                                            nullptr, M_, E_, DA_);
}
